// NeuralODERouter_31207232373065
// MI455X (gfx1250) — matrix operand order repaired and checked by reading, not compiled, not run on hardware
//
#include <hip/hip_runtime.h>
#include <hip/hip_bf16.h>
#include <math.h>

// ---------------- CDNA5 WMMA plumbing (gfx1250, wave32) ----------------
typedef __attribute__((ext_vector_type(16))) __bf16 v16bf;
typedef __attribute__((ext_vector_type(8)))  float  v8f;

// hardware f32 -> bf16 (round-nearest-even) via single v_cvt
__device__ __forceinline__ unsigned short f2bf(float x) {
    union { __bf16 h; unsigned short u; } c;
    c.h = (__bf16)x;
    return c.u;
}

// A fragment (16x32 bf16, M=lane%16): elems 0..15 = K(kbase+hh*16+e)
// Two aligned 16-byte chunks -> 2x ds_load_b128.
__device__ __forceinline__ v16bf load_fragA(const unsigned short* rowp, int kbase, int hh) {
    union { v16bf v; uint4 q[2]; } f;
    f.q[0] = *reinterpret_cast<const uint4*>(rowp + kbase + hh * 16);
    f.q[1] = *reinterpret_cast<const uint4*>(rowp + kbase + hh * 16 + 8);
    return f.v;
}
// B fragment (32x16 bf16, N=lane%16) from [N][K]-stored matrix:
// 16 contiguous, 32B-aligned K values -> 2x ds_load_b128.
__device__ __forceinline__ v16bf load_fragB(const unsigned short* nrowp, int kbase, int hh) {
    union { v16bf v; uint4 q[2]; } f;
    const uint4* p = reinterpret_cast<const uint4*>(nrowp + kbase + hh * 16);
    f.q[0] = p[0];
    f.q[1] = p[1];
    return f.v;
}

__device__ __forceinline__ v8f wmma_bf16(v16bf a, v16bf b, v8f c) {
    return __builtin_amdgcn_wmma_f32_16x16x32_bf16(false, a, false, b, (short)0, c, false, false);
}

// ---------------- problem constants ----------------
#define D_EMB   384
#define D_FF    64
#define N_TOK   1024
#define N_BATCH 64
#define N_ROWS  (N_BATCH * N_TOK)          // 65536
#define TM      32                          // ODE rows per workgroup
#define N_STEPS 10
#define DT      0.1f
#define TAU_INV 10.0f
#define OUT_XS  (N_BATCH * N_TOK * D_EMB)  // 25165824

// =====================================================================
// Kernel 1: fused RK4 neural-ODE (40 MLP evals) + score projection.
// Weights live bf16-transposed in LDS; state h and RK4 k-sum live in
// registers in WMMA C-fragment layout (6 N-tiles x 8 VGPRs per lane).
// =====================================================================
struct OdeSmem {
    alignas(32) unsigned short w1t[D_FF][D_EMB];   // W1[:384,:]^T ->  48 KB
    alignas(32) unsigned short w3t[D_EMB][D_FF];   // W3^T         ->  48 KB
    alignas(32) unsigned short w2t[D_FF][D_FF];    // W2^T         ->   8 KB
    alignas(32) unsigned short heval[TM][D_EMB];   // bf16 A-input ->  24 KB
    alignas(32) unsigned short x1[TM][D_FF];       //              ->   4 KB
    alignas(32) unsigned short x2[TM][D_FF];       //              ->   4 KB
    float bias1[D_FF];
    float b2s[D_FF];
    float b3s[D_EMB];
    float wss[D_EMB];
    float sp[4][TM];                               // score partials
};

__global__ __launch_bounds__(256)
void ode_scores_kernel(const float* __restrict__ emb,
                       const float* __restrict__ W1, const float* __restrict__ b1,
                       const float* __restrict__ W2, const float* __restrict__ b2,
                       const float* __restrict__ W3, const float* __restrict__ b3,
                       const float* __restrict__ Ws, const float* __restrict__ bs,
                       float* __restrict__ scores_out) {
    __shared__ OdeSmem sm;
    const int tid  = threadIdx.x;
    const int w    = tid >> 5;          // wave 0..7
    const int lane = tid & 31;
    const int lr   = lane & 15;         // N / M index inside tile
    const int hh   = lane >> 4;         // half-wave
    const int rowBase = blockIdx.x * TM;

    // ---- cooperative weight load (global f32 -> LDS bf16, transposed) ----
    for (int idx = tid; idx < D_EMB * D_FF; idx += 256) {          // W1[:384]
        int k = idx / D_FF, n = idx % D_FF;
        sm.w1t[n][k] = f2bf(W1[idx]);
    }
    for (int idx = tid; idx < D_FF * D_FF; idx += 256) {           // W2
        int k = idx / D_FF, n = idx % D_FF;
        sm.w2t[n][k] = f2bf(W2[idx]);
    }
    for (int idx = tid; idx < D_FF * D_EMB; idx += 256) {          // W3
        int k = idx / D_EMB, n = idx % D_EMB;
        sm.w3t[n][k] = f2bf(W3[idx]);
    }
    for (int idx = tid; idx < D_FF;  idx += 256) sm.b2s[idx] = b2[idx];
    for (int idx = tid; idx < D_EMB; idx += 256) { sm.b3s[idx] = b3[idx]; sm.wss[idx] = Ws[idx]; }

    // ---- load h0 into C-fragment-layout registers; seed heval(bf16) ----
    const int jt3 = w & 1;              // GEMM3 j-tile of this wave
    const int ntB = (w >> 1) * 6;       // GEMM3 first n-tile of this wave
    float hreg[6][8];
    float ksum[6][8];
#pragma unroll
    for (int q = 0; q < 6; ++q) {
        int ncol = (ntB + q) * 16 + lr;
#pragma unroll
        for (int v = 0; v < 8; ++v) {
            int r = jt3 * 16 + v + 8 * hh;
            float h0 = emb[(long)(rowBase + r) * D_EMB + ncol];
            hreg[q][v] = h0; ksum[q][v] = 0.f;
            sm.heval[r][ncol] = f2bf(h0);
        }
    }
    __syncthreads();

    const int jt = w & 1;               // GEMM1/2 j-tile
    const int nt = w >> 1;              // GEMM1/2 n-tile (0..3)

    for (int step = 0; step < N_STEPS; ++step) {
        float t0 = step * DT;
        for (int s = 0; s < 4; ++s) {
            float teval = t0 + ((s == 0) ? 0.f : (s == 3) ? DT : 0.5f * DT);
            float wco   = (s == 1 || s == 2) ? 2.f : 1.f;
            float aC    = (s <= 1) ? 0.5f * DT : DT;
            // time-folded bias: b1 + t * W1[384,:]
            if (tid < D_FF) sm.bias1[tid] = b1[tid] + teval * W1[D_EMB * D_FF + tid];
            __syncthreads();

            // ---- GEMM1: X1 = tanh(heval(32x384) @ W1 + bias1) ----
            {
                v8f acc = {};
                const unsigned short* arow = &sm.heval[jt * 16 + lr][0];
                const unsigned short* brow = &sm.w1t[nt * 16 + lr][0];
#pragma unroll
                for (int kk = 0; kk < D_EMB / 32; ++kk) {
                    v16bf a = load_fragA(arow, kk * 32, hh);
                    v16bf b = load_fragB(brow, kk * 32, hh);
                    acc = wmma_bf16(a, b, acc);
                }
                int ncol = nt * 16 + lr;
                float bia = sm.bias1[ncol];
#pragma unroll
                for (int v = 0; v < 8; ++v)
                    sm.x1[jt * 16 + v + 8 * hh][ncol] = f2bf(tanhf(acc[v] + bia));
            }
            __syncthreads();

            // ---- GEMM2: X2 = tanh(X1(32x64) @ W2 + b2) ----
            {
                v8f acc = {};
                const unsigned short* arow = &sm.x1[jt * 16 + lr][0];
                const unsigned short* brow = &sm.w2t[nt * 16 + lr][0];
#pragma unroll
                for (int kk = 0; kk < 2; ++kk) {
                    v16bf a = load_fragA(arow, kk * 32, hh);
                    v16bf b = load_fragB(brow, kk * 32, hh);
                    acc = wmma_bf16(a, b, acc);
                }
                int ncol = nt * 16 + lr;
                float bia = sm.b2s[ncol];
#pragma unroll
                for (int v = 0; v < 8; ++v)
                    sm.x2[jt * 16 + v + 8 * hh][ncol] = f2bf(tanhf(acc[v] + bia));
            }
            __syncthreads();

            // ---- GEMM3: k = X2(32x64) @ W3 + b3 ; RK4 update ----
            {
                v8f acc[6];
#pragma unroll
                for (int q = 0; q < 6; ++q) { v8f z = {}; acc[q] = z; }
                const unsigned short* arow = &sm.x2[jt3 * 16 + lr][0];
#pragma unroll
                for (int kk = 0; kk < 2; ++kk) {
                    v16bf a = load_fragA(arow, kk * 32, hh);
#pragma unroll
                    for (int q = 0; q < 6; ++q) {
                        const unsigned short* brow = &sm.w3t[(ntB + q) * 16 + lr][0];
                        v16bf b = load_fragB(brow, kk * 32, hh);
                        acc[q] = wmma_bf16(a, b, acc[q]);
                    }
                }
#pragma unroll
                for (int q = 0; q < 6; ++q) {
                    int ncol = (ntB + q) * 16 + lr;
                    float bia = sm.b3s[ncol];
#pragma unroll
                    for (int v = 0; v < 8; ++v) {
                        int r = jt3 * 16 + v + 8 * hh;
                        float kv = acc[q][v] + bia;
                        ksum[q][v] += wco * kv;
                        float he;
                        if (s < 3) he = hreg[q][v] + aC * kv;
                        else { hreg[q][v] += (DT / 6.0f) * ksum[q][v]; ksum[q][v] = 0.f; he = hreg[q][v]; }
                        sm.heval[r][ncol] = f2bf(he);
                    }
                }
            }
            __syncthreads();
        }
    }

    // ---- scores = h_final @ Ws + bs (deterministic shuffle+LDS reduce) ----
#pragma unroll
    for (int v = 0; v < 8; ++v) {
        float part = 0.f;
#pragma unroll
        for (int q = 0; q < 6; ++q) part += hreg[q][v] * sm.wss[(ntB + q) * 16 + lr];
        part += __shfl_xor(part, 1);
        part += __shfl_xor(part, 2);
        part += __shfl_xor(part, 4);
        part += __shfl_xor(part, 8);
        if (lr == 0) sm.sp[w >> 1][jt3 * 16 + v + 8 * hh] = part;
    }
    __syncthreads();
    if (tid < TM) {
        float sv = bs[0];
#pragma unroll
        for (int g = 0; g < 4; ++g) sv += sm.sp[g][tid];
        scores_out[rowBase + tid] = sv;
    }
}

// =====================================================================
// Kernel 2a: rowsum_i = sum_j |s_i - s_j| per batch
// =====================================================================
__global__ __launch_bounds__(256)
void rowsum_kernel(const float* __restrict__ scores, float* __restrict__ rowsum) {
    __shared__ float ss[N_TOK];
    int b = blockIdx.x;
    for (int i = threadIdx.x; i < N_TOK; i += 256) ss[i] = scores[b * N_TOK + i];
    __syncthreads();
    for (int i = threadIdx.x; i < N_TOK; i += 256) {
        float si = ss[i], acc = 0.f;
        for (int j = 0; j < N_TOK; ++j) acc += fabsf(si - ss[j]);
        rowsum[b * N_TOK + i] = acc;
    }
}

// =====================================================================
// Kernel 2b: x_sorted[b,j,:] = softmax_i((s_i*scal_j - rowsum_i)/TAU) @ emb
// P tiles generated on the fly in bf16 LDS, GEMM via v_wmma_f32_16x16x32_bf16.
// Block: one batch x 64 j-rows. 8 waves, 12 C-tiles each (64x384 output).
// =====================================================================
struct SortSmem {
    float ss[N_TOK];
    float rs[N_TOK];
    float mj[64];
    float zinv[64];
    alignas(32) unsigned short P[64][64];          // bf16 softmax tile (j x i_local)
    alignas(32) unsigned short embT[D_EMB][64];    // bf16 emb chunk, [n][i_local]
};

__global__ __launch_bounds__(256)
void softsort_kernel(const float* __restrict__ scores, const float* __restrict__ rowsum,
                     const float* __restrict__ emb, float* __restrict__ out) {
    __shared__ SortSmem sm;
    const int tid = threadIdx.x;
    const int w = tid >> 5, lane = tid & 31, lr = lane & 15, hh = lane >> 4;
    const int jb = blockIdx.x;          // 0..15 (tile of 64 j rows)
    const int b  = blockIdx.y;          // batch

    for (int i = tid; i < N_TOK; i += 256) {
        sm.ss[i] = scores[b * N_TOK + i];
        sm.rs[i] = rowsum[b * N_TOK + i];
    }
    __syncthreads();

    // per-j softmax max and 1/Z over full i-range (4 threads per j row)
    const int jl  = tid >> 2;           // 0..63
    const int seg = tid & 3;
    const int jg  = jb * 64 + jl;
    const float scal = 1023.0f - 2.0f * (float)jg;
    {
        float mx = -3.4e38f;
        for (int i = seg * 256; i < seg * 256 + 256; ++i)
            mx = fmaxf(mx, sm.ss[i] * scal - sm.rs[i]);
        mx = fmaxf(mx, __shfl_xor(mx, 1));
        mx = fmaxf(mx, __shfl_xor(mx, 2));
        float zs = 0.f;
        for (int i = seg * 256; i < seg * 256 + 256; ++i)
            zs += expf((sm.ss[i] * scal - sm.rs[i] - mx) * TAU_INV);
        zs += __shfl_xor(zs, 1);
        zs += __shfl_xor(zs, 2);
        if (seg == 0) { sm.mj[jl] = mx; sm.zinv[jl] = 1.0f / zs; }
    }

    // accumulators: wave w -> j-tile (w&3), n-tiles (w>>2)*12 .. +11
    const int jt = w & 3;
    const int ntB = (w >> 2) * 12;
    v8f acc[12];
#pragma unroll
    for (int q = 0; q < 12; ++q) { v8f z = {}; acc[q] = z; }

    for (int ic = 0; ic < N_TOK / 64; ++ic) {
        __syncthreads();                 // previous GEMM done with P/embT
        const int ibase = ic * 64;

        // load emb chunk (64 rows x 384) -> bf16 transposed LDS
        {
            const int il = tid >> 2;     // 0..63
            const float4* p = reinterpret_cast<const float4*>(
                emb + ((long)(b * N_TOK + ibase + il)) * D_EMB + seg * 96);
            if (ic + 1 < N_TOK / 64)
                __builtin_prefetch(emb + ((long)(b * N_TOK + ibase + 64 + il)) * D_EMB + seg * 96, 0, 0);
#pragma unroll
            for (int t = 0; t < 24; ++t) {
                float4 vv = p[t];
                int n0 = seg * 96 + t * 4;
                sm.embT[n0 + 0][il] = f2bf(vv.x);
                sm.embT[n0 + 1][il] = f2bf(vv.y);
                sm.embT[n0 + 2][il] = f2bf(vv.z);
                sm.embT[n0 + 3][il] = f2bf(vv.w);
            }
        }
        // P tile 64x64: p = exp((l - m)/tau) * (1/Z) folded in here
        {
            float m = sm.mj[jl], zi = sm.zinv[jl];
#pragma unroll
            for (int e = 0; e < 16; ++e) {
                int il = seg * 16 + e;
                int i  = ibase + il;
                float l = sm.ss[i] * scal - sm.rs[i];
                sm.P[jl][il] = f2bf(expf((l - m) * TAU_INV) * zi);
            }
        }
        __syncthreads();

        // GEMM: P(64x64) @ embChunk(64x384), accumulate
        const unsigned short* arow = &sm.P[jt * 16 + lr][0];
#pragma unroll
        for (int kk = 0; kk < 2; ++kk) {
            v16bf a = load_fragA(arow, kk * 32, hh);
#pragma unroll
            for (int q = 0; q < 12; ++q) {
                const unsigned short* brow = &sm.embT[(ntB + q) * 16 + lr][0];
                v16bf bfr = load_fragB(brow, kk * 32, hh);
                acc[q] = wmma_bf16(a, bfr, acc[q]);
            }
        }
    }

    // write x_sorted
#pragma unroll
    for (int q = 0; q < 12; ++q) {
        int ncol = (ntB + q) * 16 + lr;
#pragma unroll
        for (int v = 0; v < 8; ++v) {
            int jrow = jb * 64 + jt * 16 + v + 8 * hh;
            out[((long)b * N_TOK + jrow) * D_EMB + ncol] = acc[q][v];
        }
    }
}

// =====================================================================
extern "C" void kernel_launch(void* const* d_in, const int* in_sizes, int n_in,
                              void* d_out, int out_size, void* d_ws, size_t ws_size,
                              hipStream_t stream) {
    const float* emb = (const float*)d_in[0];
    const float* W1  = (const float*)d_in[1];
    const float* b1  = (const float*)d_in[2];
    const float* W2  = (const float*)d_in[3];
    const float* b2  = (const float*)d_in[4];
    const float* W3  = (const float*)d_in[5];
    const float* b3  = (const float*)d_in[6];
    const float* Ws  = (const float*)d_in[7];
    const float* bs  = (const float*)d_in[8];

    float* out    = (float*)d_out;
    float* scores = out + OUT_XS;            // scores section of d_out
    float* rowsum = (float*)d_ws;            // 64*1024 f32 scratch

    ode_scores_kernel<<<N_ROWS / TM, 256, 0, stream>>>(
        emb, W1, b1, W2, b2, W3, b3, Ws, bs, scores);
    rowsum_kernel<<<N_BATCH, 256, 0, stream>>>(scores, rowsum);
    softsort_kernel<<<dim3(N_TOK / 64, N_BATCH), 256, 0, stream>>>(
        scores, rowsum, emb, out);
}
